// Gemma2Attention_7413113553014
// MI455X (gfx1250) — matrix operand order repaired and checked by reading, not compiled, not run on hardware
//
#include <hip/hip_runtime.h>
#include <hip/hip_bf16.h>

// ---------------------------------------------------------------------------
// Gemma2 attention forward for MI455X (gfx1250, wave32, WMMA).
// All matmuls run on v_wmma_f32_16x16x32_bf16 (inputs converted f32->bf16).
// ---------------------------------------------------------------------------

typedef __bf16 bf16;
typedef __attribute__((ext_vector_type(16))) __bf16 v16bf;
typedef __attribute__((ext_vector_type(8)))  __bf16 v8bf;
typedef __attribute__((ext_vector_type(4)))  __bf16 v4bf;
typedef __attribute__((ext_vector_type(8)))  float  v8f;

#define B_  2
#define S_  2048
#define HID_ 2304
#define H_  8
#define KVH_ 4
#define D_  256
#define SCALE_ 0.0625f      // 256^-0.5
#define SOFTCAP_ 50.0f
#define NEG_ (-1.0e9f)

static __device__ inline v16bf cat16(v8bf lo, v8bf hi) {
    return __builtin_shufflevector(lo, hi, 0,1,2,3,4,5,6,7,8,9,10,11,12,13,14,15);
}
static __device__ inline v8f wmma_bf16(v16bf a, v16bf b, v8f c) {
    return __builtin_amdgcn_wmma_f32_16x16x32_bf16(false, a, false, b, (short)0, c, false, false);
}
// A fragment (16x32 bf16): lane holds row m=lane%16; k = half*8+{0..7} and 16+half*8+{0..7}
static __device__ inline v16bf frag_a(const bf16* row, int half) {
    v8bf lo = *(const v8bf*)(row + half * 8);
    v8bf hi = *(const v8bf*)(row + 16 + half * 8);
    return cat16(lo, hi);
}
// B fragment (32x16 bf16), n-major storage Bt[n][k]: lane holds col n=lane%16;
// k = half*8+{0..7} and 16+half*8+{0..7}
static __device__ inline v16bf frag_b(const bf16* row, int half) {
    v8bf lo = *(const v8bf*)(row + half * 8);
    v8bf hi = *(const v8bf*)(row + 16 + half * 8);
    return cat16(lo, hi);
}
static __device__ inline float fast_tanh(float x) {
#if __has_builtin(__builtin_amdgcn_tanhf)
    return __builtin_amdgcn_tanhf(x);   // gfx1250 V_TANH_F32 trans op
#else
    return tanhf(x);
#endif
}

// ---------------------------------------------------------------------------
// GEMM: C[M,N] = A[M,K] (f32) x W[K,N] (f32), bf16 WMMA, f32 accumulate.
// Compile-time N,K -> immediate-offset addressing. 256 threads (8 waves),
// tile 128x128, BK=32, LDS double-buffered (one barrier per K-step).
// ---------------------------------------------------------------------------
template <int N, int K>
__global__ __launch_bounds__(256) void gemm_bf16(const float* __restrict__ A,
                                                 const float* __restrict__ W,
                                                 float* __restrict__ C) {
    __shared__ __align__(16) bf16 sA[2][128 * 40];
    __shared__ __align__(16) bf16 sB[2][128 * 40];

    const int tid  = threadIdx.x;
    const int lane = tid & 31, wave = tid >> 5;
    const int wm = wave & 3, wn = wave >> 2;
    const int l15 = lane & 15, half = lane >> 4;
    const int m0 = blockIdx.y * 128, n0 = blockIdx.x * 128;

    const int arow = tid >> 3, acol = (tid & 7) * 4;   // A stage: 32 rows / pass
    const int bk   = tid >> 5, bn   = (tid & 31) * 4;  // B stage: 8 k-rows / pass

    const float* aptr = A + (size_t)(m0 + arow) * K + acol;
    const float* wptr = W + (size_t)bk * N + n0 + bn;

    v8f zero8 = {0.f,0.f,0.f,0.f,0.f,0.f,0.f,0.f};
    v8f acc[2][4];
#pragma unroll
    for (int sm = 0; sm < 2; ++sm)
#pragma unroll
        for (int sn = 0; sn < 4; ++sn) acc[sm][sn] = zero8;

    float4 av[4], wv[4];
    auto load_regs = [&](int k0) {
#pragma unroll
        for (int rr = 0; rr < 4; ++rr)
            av[rr] = *(const float4*)(aptr + (size_t)rr * 32 * K + k0);
#pragma unroll
        for (int kk = 0; kk < 4; ++kk)
            wv[kk] = *(const float4*)(wptr + (size_t)(k0 + kk * 8) * N);
    };
    auto store_lds = [&](int buf) {
#pragma unroll
        for (int rr = 0; rr < 4; ++rr) {
            v4bf p;
            p[0] = (bf16)av[rr].x; p[1] = (bf16)av[rr].y;
            p[2] = (bf16)av[rr].z; p[3] = (bf16)av[rr].w;
            *(v4bf*)(&sA[buf][(arow + rr * 32) * 40 + acol]) = p;
        }
#pragma unroll
        for (int kk = 0; kk < 4; ++kk) {
            int k = bk + kk * 8;
            sB[buf][(bn + 0) * 40 + k] = (bf16)wv[kk].x;
            sB[buf][(bn + 1) * 40 + k] = (bf16)wv[kk].y;
            sB[buf][(bn + 2) * 40 + k] = (bf16)wv[kk].z;
            sB[buf][(bn + 3) * 40 + k] = (bf16)wv[kk].w;
        }
    };

    load_regs(0);
    store_lds(0);
    __syncthreads();

    int buf = 0;
    for (int k0 = 0; k0 < K; k0 += 32) {
        const bool has_next = (k0 + 32 < K);
        if (has_next) load_regs(k0 + 32);   // global loads overlap WMMAs below

        v16bf af[2], bfv[4];
#pragma unroll
        for (int sm = 0; sm < 2; ++sm)
            af[sm] = frag_a(&sA[buf][(wm * 32 + sm * 16 + l15) * 40], half);
#pragma unroll
        for (int sn = 0; sn < 4; ++sn)
            bfv[sn] = frag_b(&sB[buf][(wn * 64 + sn * 16 + l15) * 40], half);
#pragma unroll
        for (int sm = 0; sm < 2; ++sm)
#pragma unroll
            for (int sn = 0; sn < 4; ++sn)
                acc[sm][sn] = wmma_bf16(af[sm], bfv[sn], acc[sm][sn]);

        if (has_next) store_lds(buf ^ 1);
        __syncthreads();
        buf ^= 1;
    }

    // epilogue: single base pointer, compile-time strides -> immediate offsets
    float* cp = C + (size_t)(m0 + wm * 32 + 8 * half) * N + n0 + wn * 64 + l15;
#pragma unroll
    for (int sm = 0; sm < 2; ++sm)
#pragma unroll
        for (int sn = 0; sn < 4; ++sn)
#pragma unroll
            for (int r = 0; r < 8; ++r)
                cp[(size_t)(sm * 16 + r) * N + sn * 16] = acc[sm][sn][r];
}

// ---------------------------------------------------------------------------
// RoPE + relayout (B,S,NH,D) f32 -> (B,NH,S,D) bf16.
// ---------------------------------------------------------------------------
__global__ __launch_bounds__(256) void rope_relayout(const float* __restrict__ x,
                                                     const float* __restrict__ cb,
                                                     const float* __restrict__ sb,
                                                     bf16* __restrict__ out, int NH) {
    int idx = blockIdx.x * 256 + threadIdx.x;
    int d  = idx & 127;
    int rem = idx >> 7;
    int hh = rem % NH;
    int r2 = rem / NH;
    int s  = r2 & (S_ - 1);
    int b  = r2 >> 11;
    size_t bi = ((size_t)(b * S_ + s) * NH + hh) * D_;
    float x1 = x[bi + d], x2 = x[bi + d + 128];
    size_t ci = (size_t)(b * S_ + s) * D_;
    float c1 = cb[ci + d],       s1 = sb[ci + d];
    float c2 = cb[ci + d + 128], s2 = sb[ci + d + 128];
    size_t oi = ((size_t)(b * NH + hh) * S_ + s) * D_;
    out[oi + d]       = (bf16)(x1 * c1 - x2 * s1);
    out[oi + d + 128] = (bf16)(x2 * c2 + x1 * s2);
}

// (B,S,KVH,D) f32 -> (B,KVH,S,D) bf16
__global__ __launch_bounds__(256) void v_relayout(const float* __restrict__ x,
                                                  bf16* __restrict__ out) {
    int idx = blockIdx.x * 256 + threadIdx.x;
    int d  = idx & 255;
    int rem = idx >> 8;
    int hh = rem & (KVH_ - 1);
    int r2 = rem >> 2;
    int s  = r2 & (S_ - 1);
    int b  = r2 >> 11;
    out[(((size_t)b * KVH_ + hh) * S_ + s) * D_ + d] =
        (bf16)x[(((size_t)b * S_ + s) * KVH_ + hh) * D_ + d];
}

// ---------------------------------------------------------------------------
// Flash attention w/ tanh softcap + causal mask. 128 threads (4 waves),
// 64 q-rows per block. Q staged once, K/V^T staged per 64-wide tile in LDS
// (~111 KB of the WGP's 320 KB LDS), so steady-state global traffic is just
// one 32 KB K tile + one 32 KB V tile per iteration (all L2-resident).
// ---------------------------------------------------------------------------
__global__ __launch_bounds__(128) void flash_attn(const bf16* __restrict__ qb,
                                                  const bf16* __restrict__ kb,
                                                  const bf16* __restrict__ vb,
                                                  float* __restrict__ attn) {
    __shared__ __align__(16) bf16 sQ[64 * 264];   // [qrow][d], pitch 264 (528B)
    __shared__ __align__(16) bf16 sK[64 * 264];   // [krow][d], pitch 264
    __shared__ __align__(16) bf16 sVt[256 * 72];  // V^T: [d][k], pitch 72 (144B)
    __shared__ __align__(16) bf16 sP[64 * 72];    // P bf16, pitch 72

    const int tid  = threadIdx.x;
    const int lane = tid & 31, w = tid >> 5;
    const int l15 = lane & 15, half = lane >> 4;
    const int q0 = blockIdx.x * 64;
    const int h  = blockIdx.y;
    const int b  = blockIdx.z;
    const int kvh = h >> 1;  // N_REP = 2

    const bf16* qblk  = qb + (((size_t)b * H_ + h) * S_ + q0) * D_;
    const bf16* kbase = kb + (((size_t)b * KVH_ + kvh) * S_) * D_;
    const bf16* vbase = vb + (((size_t)b * KVH_ + kvh) * S_) * D_;

    // stage Q tile once (64 x 256 bf16)
    for (int i = 0; i < 16; ++i) {
        int idx = i * 128 + tid;
        int qr = idx >> 5;            // 0..63
        int d8 = (idx & 31) * 8;      // 0..248
        *(v8bf*)(&sQ[qr * 264 + d8]) = *(const v8bf*)(qblk + (size_t)qr * D_ + d8);
    }

    v8f zero8 = {0.f,0.f,0.f,0.f,0.f,0.f,0.f,0.f};
    v8f o[16];
#pragma unroll
    for (int nt = 0; nt < 16; ++nt) o[nt] = zero8;
    float rm[8], rl[8];
#pragma unroll
    for (int r = 0; r < 8; ++r) { rm[r] = -1.0e30f; rl[r] = 0.f; }

    const int nkt = (q0 >> 6) + 1;  // causal: only tiles with k0 <= q0
    for (int kt = 0; kt < nkt; ++kt) {
        const int k0 = kt * 64;
        __syncthreads();  // prior reads of sK/sVt done (also orders sQ staging)
        // stage K tile (row-major) and V^T tile
        for (int i = 0; i < 16; ++i) {
            int idx = i * 128 + tid;
            int kr = idx >> 5;
            int d8 = (idx & 31) * 8;
            *(v8bf*)(&sK[kr * 264 + d8]) =
                *(const v8bf*)(kbase + (size_t)(k0 + kr) * D_ + d8);
            v8bf vv = *(const v8bf*)(vbase + (size_t)(k0 + kr) * D_ + d8);
#pragma unroll
            for (int j = 0; j < 8; ++j) sVt[(d8 + j) * 72 + kr] = vv[j];
        }
        __syncthreads();

        // ---- S = Q K^T : 8 k-chunks of 32 over D, 4 n-tiles of 16 kv rows
        v8f s[4];
#pragma unroll
        for (int t = 0; t < 4; ++t) s[t] = zero8;
#pragma unroll
        for (int c = 0; c < 8; ++c) {
            v16bf aq = frag_a(&sQ[(w * 16 + l15) * 264 + c * 32], half);
#pragma unroll
            for (int t = 0; t < 4; ++t)
                s[t] = wmma_bf16(aq, frag_b(&sK[(t * 16 + l15) * 264 + c * 32], half), s[t]);
        }

        // ---- softcap + causal mask + online softmax; P -> LDS as bf16
#pragma unroll
        for (int r = 0; r < 8; ++r) {
            const int qidx = q0 + w * 16 + r + 8 * half;
            float vals[4];
            float mx = -1.0e30f;
#pragma unroll
            for (int t = 0; t < 4; ++t) {
                float v = s[t][r] * SCALE_;
                v = fast_tanh(v * (1.0f / SOFTCAP_)) * SOFTCAP_;
                int kidx = k0 + t * 16 + l15;
                if (kidx > qidx) v = NEG_;
                vals[t] = v;
                mx = fmaxf(mx, v);
            }
            mx = fmaxf(mx, __shfl_xor(mx, 1, 32));
            mx = fmaxf(mx, __shfl_xor(mx, 2, 32));
            mx = fmaxf(mx, __shfl_xor(mx, 4, 32));
            mx = fmaxf(mx, __shfl_xor(mx, 8, 32));
            float nm = fmaxf(rm[r], mx);
            float sc = __expf(rm[r] - nm);
            rm[r] = nm;
            float sum = 0.f;
#pragma unroll
            for (int t = 0; t < 4; ++t) {
                float p = __expf(vals[t] - nm);
                sum += p;
                sP[(w * 16 + r + 8 * half) * 72 + t * 16 + l15] = (bf16)p;
            }
            sum += __shfl_xor(sum, 1, 32);
            sum += __shfl_xor(sum, 2, 32);
            sum += __shfl_xor(sum, 4, 32);
            sum += __shfl_xor(sum, 8, 32);
            rl[r] = rl[r] * sc + sum;
#pragma unroll
            for (int nt = 0; nt < 16; ++nt) o[nt][r] *= sc;
        }

        // ---- O += P V : A = P (16x64 per wave, 2 k-chunks), B = V^T from LDS
        // (sP written/read by the same wave: DS ops are in-order per wave)
#pragma unroll
        for (int kc = 0; kc < 2; ++kc) {
            v16bf ap = frag_a(&sP[(w * 16 + l15) * 72 + kc * 32], half);
#pragma unroll
            for (int nt = 0; nt < 16; ++nt)
                o[nt] = wmma_bf16(ap, frag_b(&sVt[(nt * 16 + l15) * 72 + kc * 32], half), o[nt]);
        }
    }

    // ---- normalize & write (B*S, H*D) with col = h*256 + d
#pragma unroll
    for (int r = 0; r < 8; ++r) {
        float inv = 1.0f / rl[r];
        size_t row = (size_t)b * S_ + q0 + w * 16 + r + 8 * half;
        float* out = attn + row * (H_ * D_) + h * D_;
#pragma unroll
        for (int nt = 0; nt < 16; ++nt)
            out[nt * 16 + l15] = o[nt][r] * inv;
    }
}

// ---------------------------------------------------------------------------
extern "C" void kernel_launch(void* const* d_in, const int* in_sizes, int n_in,
                              void* d_out, int out_size, void* d_ws, size_t ws_size,
                              hipStream_t stream) {
    const float* X    = (const float*)d_in[0];
    const float* cosb = (const float*)d_in[1];
    const float* sinb = (const float*)d_in[2];
    // d_in[3] = attention_mask: causal mask computed in-kernel
    const float* Wq = (const float*)d_in[4];
    const float* Wk = (const float*)d_in[5];
    const float* Wv = (const float*)d_in[6];
    const float* Wo = (const float*)d_in[7];
    float* out = (float*)d_out;

    char* ws = (char*)d_ws;                 // needs ~96 MB
    float* qf = (float*)(ws);               // 32 MB (B*S x H*D f32); reused as attn
    float* kf = (float*)(ws + 33554432);    // 16 MB
    float* vf = (float*)(ws + 50331648);    // 16 MB
    bf16*  qb = (bf16*)(ws + 67108864);     // 16 MB (B,H,S,D) bf16
    bf16*  kb = (bf16*)(ws + 83886080);     //  8 MB (B,KVH,S,D) bf16
    bf16*  vb = (bf16*)(ws + 92274688);     //  8 MB (B,KVH,S,D) bf16
    float* attn = qf;                       // alias: qf dead after rope

    const int M = B_ * S_;  // 4096
    gemm_bf16<H_ * D_, HID_><<<dim3((H_ * D_) / 128, M / 128), 256, 0, stream>>>(X, Wq, qf);
    gemm_bf16<KVH_ * D_, HID_><<<dim3((KVH_ * D_) / 128, M / 128), 256, 0, stream>>>(X, Wk, kf);
    gemm_bf16<KVH_ * D_, HID_><<<dim3((KVH_ * D_) / 128, M / 128), 256, 0, stream>>>(X, Wv, vf);

    rope_relayout<<<(B_ * S_ * H_ * 128) / 256, 256, 0, stream>>>(qf, cosb, sinb, qb, H_);
    rope_relayout<<<(B_ * S_ * KVH_ * 128) / 256, 256, 0, stream>>>(kf, cosb, sinb, kb, KVH_);
    v_relayout<<<(B_ * S_ * KVH_ * D_) / 256, 256, 0, stream>>>(vf, vb);

    flash_attn<<<dim3(S_ / 64, H_, B_), 128, 0, stream>>>(qb, kb, vb, attn);

    gemm_bf16<HID_, H_ * D_><<<dim3(HID_ / 128, M / 128), 256, 0, stream>>>(attn, Wo, out);
}
